// ParallelMoELayer_53068615909668
// MI455X (gfx1250) — matrix operand order repaired and checked by reading, not compiled, not run on hardware
//
#include <hip/hip_runtime.h>
#include <hip/hip_bf16.h>

// ---------------------------------------------------------------------------
// MoE (top-2, 8 experts) for MI455X / gfx1250.
//   T=8192 tokens, D=1024, I=2048.  Effective FLOPs (gathered) ~206 GFLOP.
//   bf16 weights (~100MB) fit in the 192MB L2 -> expert GEMMs are L2-resident.
//   MT=32 tokens/block halves weight traffic vs MT=16 (each B fragment feeds
//   two M-slices).  Matrix path: v_wmma_f32_16x16x32_bf16, fp32 accumulate.
// ---------------------------------------------------------------------------

#define T_TOKENS 8192
#define D_MODEL  1024
#define INTER    2048
#define NEXP     8
#define MT       32          // tokens per tile (two 16-row M-slices)
#define ICHUNK   128         // intermediate-dim chunk
#define XS_STRIDE 1032       // padded LDS stride (halves) -> conflict free
#define GS_STRIDE 132        // padded f32 stride
#define HS_STRIDE 136        // padded bf16 stride
#define MAX_TILES (T_TOKENS * 2 / MT + NEXP)   // 520 worst case

#define XS_BYTES (MT * XS_STRIDE * 2)                 // 66048
#define GS_BYTES (MT * GS_STRIDE * 4)                 // 16896
#define HS_BYTES (MT * HS_STRIDE * 2)                 // 8704
#define OFF_GS   (XS_BYTES)
#define OFF_US   (OFF_GS + GS_BYTES)
#define OFF_HS   (OFF_US + GS_BYTES)
#define OFF_TOK  (OFF_HS + HS_BYTES)
#define OFF_W    (OFF_TOK + MT * 4)
#define SMEM_TOTAL (OFF_W + MT * 4)                   // 108800 bytes (< 320KB WGP LDS)

typedef __bf16 bf16_t;
typedef __attribute__((ext_vector_type(16))) __bf16 v16bf;
typedef __attribute__((ext_vector_type(8)))  __bf16 v8bf;
typedef __attribute__((ext_vector_type(8)))  float  v8f;

__device__ __forceinline__ bf16_t f2bf(float f) {
  unsigned u = __builtin_bit_cast(unsigned, f);
  unsigned r = u + 0x7FFFu + ((u >> 16) & 1u);   // round to nearest even
  unsigned short h = (unsigned short)(r >> 16);
  return __builtin_bit_cast(bf16_t, h);
}

// A fragment (16x32 bf16, M x K): lanes 0-15 hold K{0..7,16..23}, lanes 16-31
// hold K{8..15,24..31} for row m = lane%16 (CDNA5 ISA 16-bit A layout).
__device__ __forceinline__ v16bf load_a_lds(const bf16_t* base, int stride, int k0, int lane) {
  int m  = lane & 15;
  int kh = (lane < 16) ? 0 : 8;
  const bf16_t* p = base + m * stride + k0 + kh;
  v8bf lo = *(const v8bf*)(p);
  v8bf hi = *(const v8bf*)(p + 16);
  v16bf a;
#pragma unroll
  for (int i = 0; i < 8; ++i) { a[i] = lo[i]; a[i + 8] = hi[i]; }
  return a;
}

// B fragment (32x16 bf16, K x N): lanes 0-15 hold K{0..7,16..23} of column n=lane,
// lanes 16-31 hold K{8..15,24..31}.  Weight rows are K-contiguous -> two 16B loads.
__device__ __forceinline__ v16bf load_b_glb(const bf16_t* W, int rowStride, int n0, int k0, int lane) {
  int n  = lane & 15;
  int kk = (lane < 16) ? 0 : 8;
  const bf16_t* p = W + (size_t)(n0 + n) * rowStride + (k0 + kk);
  v8bf lo = *(const v8bf*)(p);
  v8bf hi = *(const v8bf*)(p + 16);
  v16bf b;
#pragma unroll
  for (int i = 0; i < 8; ++i) { b[i] = lo[i]; b[i + 8] = hi[i]; }
  return b;
}

__device__ __forceinline__ v8f wmma_bf16(v16bf a, v16bf b, v8f c) {
  return __builtin_amdgcn_wmma_f32_16x16x32_bf16(false, a, false, b, (short)0, c, false, false);
}

// ---------------------------------------------------------------------------
__global__ void zero_f32_x4(float* __restrict__ p, int n4) {
  int i = blockIdx.x * blockDim.x + threadIdx.x;
  if (i < n4) ((float4*)p)[i] = make_float4(0.f, 0.f, 0.f, 0.f);
}

__global__ void cvt_f32_bf16_x4(const float* __restrict__ src, bf16_t* __restrict__ dst, int n4) {
  int i = blockIdx.x * blockDim.x + threadIdx.x;
  if (i >= n4) return;
  float4 f = ((const float4*)src)[i];
  ushort4 h;
  h.x = __builtin_bit_cast(unsigned short, f2bf(f.x));
  h.y = __builtin_bit_cast(unsigned short, f2bf(f.y));
  h.z = __builtin_bit_cast(unsigned short, f2bf(f.z));
  h.w = __builtin_bit_cast(unsigned short, f2bf(f.w));
  ((ushort4*)dst)[i] = h;
}

// One wave per token: logits, softmax, top-2, renormalize.
__global__ __launch_bounds__(256) void router_kernel(
    const float* __restrict__ x, const float* __restrict__ rw,
    int* __restrict__ tIdx, float* __restrict__ tW) {
  int wv = threadIdx.x >> 5, lane = threadIdx.x & 31;
  int t = blockIdx.x * 8 + wv;
  if (t >= T_TOKENS) return;
  const float* xr = x + (size_t)t * D_MODEL;
  float acc[NEXP];
#pragma unroll
  for (int e = 0; e < NEXP; ++e) acc[e] = 0.f;
  for (int d = lane; d < D_MODEL; d += 32) {
    float xv = xr[d];
#pragma unroll
    for (int e = 0; e < NEXP; ++e) acc[e] += xv * rw[e * D_MODEL + d];
  }
#pragma unroll
  for (int e = 0; e < NEXP; ++e)
    for (int off = 16; off > 0; off >>= 1) acc[e] += __shfl_xor(acc[e], off, 32);
  if (lane == 0) {
    float mx = acc[0];
#pragma unroll
    for (int e = 1; e < NEXP; ++e) mx = fmaxf(mx, acc[e]);
    float p[NEXP], s = 0.f;
#pragma unroll
    for (int e = 0; e < NEXP; ++e) { p[e] = __expf(acc[e] - mx); s += p[e]; }
#pragma unroll
    for (int e = 0; e < NEXP; ++e) p[e] /= s;
    int i0 = 0;
#pragma unroll
    for (int e = 1; e < NEXP; ++e) if (p[e] > p[i0]) i0 = e;
    int i1 = (i0 == 0) ? 1 : 0;
#pragma unroll
    for (int e = 0; e < NEXP; ++e) if (e != i0 && p[e] > p[i1]) i1 = e;
    float ws = p[i0] + p[i1];
    tIdx[t * 2 + 0] = i0;  tW[t * 2 + 0] = p[i0] / ws;
    tIdx[t * 2 + 1] = i1;  tW[t * 2 + 1] = p[i1] / ws;
  }
}

// Single block: counts -> offsets -> scatter (token,weight) into expert segments.
__global__ __launch_bounds__(256) void build_segments(
    const int* __restrict__ tIdx, const float* __restrict__ tW,
    int* __restrict__ segTok, float* __restrict__ segW,
    int* __restrict__ segOff, int* __restrict__ tileOff) {
  __shared__ int cnt[NEXP];
  __shared__ int pos[NEXP];
  __shared__ int soff[NEXP + 1];
  int tid = threadIdx.x;
  if (tid < NEXP) cnt[tid] = 0;
  __syncthreads();
  for (int p = tid; p < T_TOKENS * 2; p += 256) atomicAdd(&cnt[tIdx[p]], 1);
  __syncthreads();
  if (tid == 0) {
    int a = 0, ta = 0;
    for (int e = 0; e < NEXP; ++e) {
      soff[e] = a; segOff[e] = a; tileOff[e] = ta;
      ta += (cnt[e] + MT - 1) / MT;
      a += cnt[e];
    }
    soff[NEXP] = a; segOff[NEXP] = a; tileOff[NEXP] = ta;
  }
  __syncthreads();
  if (tid < NEXP) pos[tid] = soff[tid];
  __syncthreads();
  for (int p = tid; p < T_TOKENS * 2; p += 256) {
    int e = tIdx[p];
    int slot = atomicAdd(&pos[e], 1);
    segTok[slot] = p >> 1;
    segW[slot]   = tW[p];
  }
}

// ---------------------------------------------------------------------------
// Fused expert kernel: per block = (expert, tile of 32 gathered tokens).
// 8 waves, dynamic LDS (~109KB).  Per 128-wide I-chunk:
//   waves 0-3 -> G, waves 4-7 -> U: each wave owns 32 chunk columns and both
//   16-row M-slices (4 WMMAs per k-step against 2 B loads),
//   silu(G)*U -> bf16 H in LDS,
//   down-projection: each wave owns 128 output cols x both M-slices
//   (16 WMMAs per k-step against 8 B loads), Y held in 128 VGPRs/lane.
// Software-pipelined (next k-step's A/B fragments preloaded).
// ---------------------------------------------------------------------------
__global__ __launch_bounds__(256) void moe_expert_gemm(
    float* __restrict__ out,
    const bf16_t* __restrict__ Xbf,
    const bf16_t* __restrict__ Gbf,
    const bf16_t* __restrict__ Ubf,
    const bf16_t* __restrict__ Dbf,
    const int* __restrict__ segTok,
    const float* __restrict__ segW,
    const int* __restrict__ segOff,
    const int* __restrict__ tileOff) {
  extern __shared__ __align__(16) char smem[];
  bf16_t* Xs  = (bf16_t*)smem;                    // [MT][XS_STRIDE]
  float*  Gs  = (float*)(smem + OFF_GS);          // [MT][GS_STRIDE]
  float*  Us  = (float*)(smem + OFF_US);          // [MT][GS_STRIDE]
  bf16_t* Hs  = (bf16_t*)(smem + OFF_HS);         // [MT][HS_STRIDE]
  int*    sTok = (int*)(smem + OFF_TOK);
  float*  sW   = (float*)(smem + OFF_W);

  int b = blockIdx.x;
  if (b >= tileOff[NEXP]) return;
  int e = 0;
  while (e < NEXP - 1 && b >= tileOff[e + 1]) ++e;
  int tile = b - tileOff[e];
  int segStart = segOff[e] + tile * MT;
  int cnt = segOff[e + 1] - segStart;
  if (cnt > MT) cnt = MT;

  int tid = threadIdx.x;
  if (tid < MT) {
    if (tid < cnt) { sTok[tid] = segTok[segStart + tid]; sW[tid] = segW[segStart + tid]; }
    else           { sTok[tid] = segTok[segStart];       sW[tid] = 0.f; }   // padded slot: weight 0
  }
  __syncthreads();

  // Stage gathered X rows (bf16) into LDS: 32 rows x 1024 halves.
  {
    int m = tid >> 3, j = tid & 7;
    const bf16_t* src = Xbf + (size_t)sTok[m] * D_MODEL;
#pragma unroll
    for (int it = 0; it < 16; ++it) {
      int colh = (it * 8 + j) * 8;
      *(uint4*)&Xs[m * XS_STRIDE + colh] = *(const uint4*)(src + colh);
    }
  }
  __syncthreads();

  int wv = tid >> 5, lane = tid & 31;
  const bf16_t* Wg = Gbf + (size_t)e * INTER * D_MODEL;
  const bf16_t* Wu = Ubf + (size_t)e * INTER * D_MODEL;
  const bf16_t* Wd = Dbf + (size_t)e * D_MODEL * INTER;
  const bf16_t* Wgu = (wv < 4) ? Wg : Wu;
  float* GUs = (wv < 4) ? Gs : Us;
  int wq = wv & 3;
  const bf16_t* Xs1 = Xs + 16 * XS_STRIDE;   // M-slice 1 (rows 16..31)
  const bf16_t* Hs1 = Hs + 16 * HS_STRIDE;

  const v8f vzero = {0.f, 0.f, 0.f, 0.f, 0.f, 0.f, 0.f, 0.f};
  v8f Y0[8], Y1[8];
#pragma unroll
  for (int i = 0; i < 8; ++i) { Y0[i] = vzero; Y1[i] = vzero; }

  for (int ic = 0; ic < INTER; ic += ICHUNK) {
    // ---- gate/up chunk: 2 M-slices x 2 column tiles, K=1024, pipelined ----
    v8f g00 = vzero, g01 = vzero, g10 = vzero, g11 = vzero;
    int n0 = ic + wq * 32;
    v16bf a0 = load_a_lds(Xs,  XS_STRIDE, 0, lane);
    v16bf a1 = load_a_lds(Xs1, XS_STRIDE, 0, lane);
    v16bf b0 = load_b_glb(Wgu, D_MODEL, n0,      0, lane);
    v16bf b1 = load_b_glb(Wgu, D_MODEL, n0 + 16, 0, lane);
#pragma unroll 4
    for (int k0 = 0; k0 < D_MODEL - 32; k0 += 32) {
      v16bf ca0 = a0, ca1 = a1, cb0 = b0, cb1 = b1;
      int kn = k0 + 32;
      a0 = load_a_lds(Xs,  XS_STRIDE, kn, lane);
      a1 = load_a_lds(Xs1, XS_STRIDE, kn, lane);
      b0 = load_b_glb(Wgu, D_MODEL, n0,      kn, lane);
      b1 = load_b_glb(Wgu, D_MODEL, n0 + 16, kn, lane);
      g00 = wmma_bf16(ca0, cb0, g00);
      g01 = wmma_bf16(ca0, cb1, g01);
      g10 = wmma_bf16(ca1, cb0, g10);
      g11 = wmma_bf16(ca1, cb1, g11);
    }
    g00 = wmma_bf16(a0, b0, g00);
    g01 = wmma_bf16(a0, b1, g01);
    g10 = wmma_bf16(a1, b0, g10);
    g11 = wmma_bf16(a1, b1, g11);
    {
      int colb = wq * 32 + (lane & 15);
      int mb = (lane < 16) ? 0 : 8;
#pragma unroll
      for (int r = 0; r < 8; ++r) {
        GUs[(mb + r) * GS_STRIDE + colb]             = g00[r];
        GUs[(mb + r) * GS_STRIDE + colb + 16]        = g01[r];
        GUs[(16 + mb + r) * GS_STRIDE + colb]        = g10[r];
        GUs[(16 + mb + r) * GS_STRIDE + colb + 16]   = g11[r];
      }
    }
    __syncthreads();
    // ---- H = silu(G) * U -> bf16 (each thread 16 contiguous elements) ----
    {
      int m = tid >> 3;
      int n = (tid & 7) * 16;
#pragma unroll
      for (int half = 0; half < 2; ++half) {
        v8bf hv;
#pragma unroll
        for (int jj = 0; jj < 8; ++jj) {
          float g = Gs[m * GS_STRIDE + n + half * 8 + jj];
          float u = Us[m * GS_STRIDE + n + half * 8 + jj];
          float h = (g / (1.f + __expf(-g))) * u;
          hv[jj] = f2bf(h);
        }
        *(v8bf*)&Hs[m * HS_STRIDE + n + half * 8] = hv;
      }
    }
    __syncthreads();
    // ---- down chunk: Y[32, wv*128 .. +128) += H @ down[:, ic:ic+128].T ----
    int nb = wv * 128;
#pragma unroll
    for (int ks = 0; ks < 4; ++ks) {
      int kk = ic + ks * 32;
      v16bf ha0 = load_a_lds(Hs,  HS_STRIDE, ks * 32, lane);
      v16bf ha1 = load_a_lds(Hs1, HS_STRIDE, ks * 32, lane);
      v16bf bb = load_b_glb(Wd, INTER, nb, kk, lane);
#pragma unroll
      for (int nt = 0; nt < 8; ++nt) {
        v16bf cur = bb;
        if (nt < 7) bb = load_b_glb(Wd, INTER, nb + (nt + 1) * 16, kk, lane);
        Y0[nt] = wmma_bf16(ha0, cur, Y0[nt]);
        Y1[nt] = wmma_bf16(ha1, cur, Y1[nt]);
      }
    }
    // No barrier needed here: next chunk's Gs/Us/Hs writes are separated from
    // this chunk's readers by the two barriers above.
  }

  // ---- scatter-add we * y (exactly 2 commutative adds per output element) ----
  {
    int nloc = lane & 15;
    int mb = (lane < 16) ? 0 : 8;
#pragma unroll
    for (int nt = 0; nt < 8; ++nt) {
      int col = wv * 128 + nt * 16 + nloc;
#pragma unroll
      for (int r = 0; r < 8; ++r) {
        int m0 = mb + r;
        int m1 = 16 + mb + r;
        atomicAdd(out + (size_t)sTok[m0] * D_MODEL + col, sW[m0] * Y0[nt][r]);
        atomicAdd(out + (size_t)sTok[m1] * D_MODEL + col, sW[m1] * Y1[nt][r]);
      }
    }
  }
}

// ---------------------------------------------------------------------------
extern "C" void kernel_launch(void* const* d_in, const int* in_sizes, int n_in,
                              void* d_out, int out_size, void* d_ws, size_t ws_size,
                              hipStream_t stream) {
  const float* x  = (const float*)d_in[0];   // [4,2048,1024]
  const float* rw = (const float*)d_in[1];   // [8,1024]
  const float* gw = (const float*)d_in[2];   // [8,2048,1024]
  const float* uw = (const float*)d_in[3];   // [8,2048,1024]
  const float* dw = (const float*)d_in[4];   // [8,1024,2048]
  float* out = (float*)d_out;

  char* ws = (char*)d_ws;
  size_t off = 0;
  auto take = [&](size_t bytes) -> void* {
    void* p = ws + off;
    off = (off + bytes + 255) & ~(size_t)255;
    return p;
  };
  bf16_t* Xbf = (bf16_t*)take((size_t)T_TOKENS * D_MODEL * 2);
  bf16_t* Gbf = (bf16_t*)take((size_t)NEXP * INTER * D_MODEL * 2);
  bf16_t* Ubf = (bf16_t*)take((size_t)NEXP * INTER * D_MODEL * 2);
  bf16_t* Dbf = (bf16_t*)take((size_t)NEXP * D_MODEL * INTER * 2);
  int*    tIdx   = (int*)  take((size_t)T_TOKENS * 2 * 4);
  float*  tW     = (float*)take((size_t)T_TOKENS * 2 * 4);
  int*    segTok = (int*)  take((size_t)T_TOKENS * 2 * 4);
  float*  segW   = (float*)take((size_t)T_TOKENS * 2 * 4);
  int*    segOff = (int*)  take((NEXP + 1) * 4);
  int*    tileOff= (int*)  take((NEXP + 1) * 4);

  int n4;
  n4 = T_TOKENS * D_MODEL / 4;
  zero_f32_x4<<<(n4 + 255) / 256, 256, 0, stream>>>(out, n4);
  cvt_f32_bf16_x4<<<(n4 + 255) / 256, 256, 0, stream>>>(x, Xbf, n4);
  n4 = NEXP * INTER * D_MODEL / 4;
  cvt_f32_bf16_x4<<<(n4 + 255) / 256, 256, 0, stream>>>(gw, Gbf, n4);
  cvt_f32_bf16_x4<<<(n4 + 255) / 256, 256, 0, stream>>>(uw, Ubf, n4);
  cvt_f32_bf16_x4<<<(n4 + 255) / 256, 256, 0, stream>>>(dw, Dbf, n4);

  router_kernel<<<T_TOKENS / 8, 256, 0, stream>>>(x, rw, tIdx, tW);
  build_segments<<<1, 256, 0, stream>>>(tIdx, tW, segTok, segW, segOff, tileOff);
  moe_expert_gemm<<<MAX_TILES, 256, SMEM_TOTAL, stream>>>(out, Xbf, Gbf, Ubf, Dbf,
                                                          segTok, segW, segOff, tileOff);
}
